// PointTransformerConv_64707977282152
// MI455X (gfx1250) — matrix operand order repaired and checked by reading, not compiled, not run on hardware
//
#include <hip/hip_runtime.h>

#define CCH 128   // channel count

typedef __attribute__((ext_vector_type(16))) _Float16 v16h;
typedef __attribute__((ext_vector_type(8)))  _Float16 v8h;
typedef __attribute__((ext_vector_type(8)))  float    v8f;

#ifndef __has_builtin
#define __has_builtin(x) 0
#endif
#if __has_builtin(__builtin_amdgcn_global_load_async_to_lds_b128) && \
    __has_builtin(__builtin_amdgcn_s_wait_asynccnt)
#define USE_ASYNC_LDS 1
#else
#define USE_ASYNC_LDS 0
#endif

// ---------------------------------------------------------------- WMMA helpers

__device__ __forceinline__ v8f wmma_f16(v16h a, v16h b, v8f c) {
  // v_wmma_f32_16x16x32_f16: (neg_a, A, neg_b, B, c_mod, C, reuse_a, reuse_b)
  return __builtin_amdgcn_wmma_f32_16x16x32_f16(false, a, false, b, (short)0, c,
                                                false, false);
}

// A-fragment (16x32 f16, MxK) from a row-major [16][128] f16 buffer in LDS.
// Contiguous k order (B uses the same): lanes 0-15 hold M=lane, K = kt*32 + {0..15};
//             lanes 16-31 hold M=lane-16, K = kt*32 + {16..31}.
__device__ __forceinline__ v16h ldsA_frag(const _Float16* base, int lane, int kt) {
  const int m  = lane & 15;
  const int kb = kt * 32 + ((lane & 16) ? 16 : 0);
  const _Float16* p = base + m * CCH + kb;
  union { v16h v; v8h h[2]; } u;
  u.h[0] = *(const v8h*)(p);        // 16B ds_load
  u.h[1] = *(const v8h*)(p + 8);    // 16B ds_load
  return u.v;
}

// B-fragment (32x16 f16, KxN) from a pre-transposed weight matrix wt[n][k]
// (contiguous in k). Lane L holds column n = L%16, K = kt*32 + (L<16?0..15:16..31).
__device__ __forceinline__ v16h glbB_frag(const _Float16* wt, int lane, int nt, int kt) {
  const int n  = nt * 16 + (lane & 15);
  const int kb = kt * 32 + ((lane & 16) ? 16 : 0);
  return *(const v16h*)(wt + (size_t)n * CCH + kb);   // 32B contiguous load
}

// Same B-fragment but from an LDS copy of the weight matrix.
__device__ __forceinline__ v16h ldsB_frag(const _Float16* wlds, int lane, int nt, int kt) {
  const int n  = nt * 16 + (lane & 15);
  const int kb = kt * 32 + ((lane & 16) ? 16 : 0);
  const _Float16* p = wlds + n * CCH + kb;
  union { v16h v; v8h h[2]; } u;
  u.h[0] = *(const v8h*)(p);
  u.h[1] = *(const v8h*)(p + 8);
  return u.v;
}

// ------------------------------------------- async weight staging global->LDS

// Issue copy of one 128x128 f16 matrix (32 KB) into LDS. With CDNA5 async-LDS
// this overlaps with subsequent VALU work until stage_wait().
__device__ __forceinline__ void stage_issue(_Float16* dst, const _Float16* src,
                                            int tid, int nthr) {
#if USE_ASYNC_LDS
  typedef int b128 __attribute__((vector_size(16)));
  b128 __attribute__((address_space(1)))* gp =
      (b128 __attribute__((address_space(1)))*)(void*)const_cast<_Float16*>(src);
  b128 __attribute__((address_space(3)))* lp =
      (b128 __attribute__((address_space(3)))*)(void*)dst;
  const int nvec = CCH * CCH * 2 / 16;   // 2048 x 16B
  for (int i = tid; i < nvec; i += nthr)
    __builtin_amdgcn_global_load_async_to_lds_b128(gp + i, lp + i, 0, 0);
#else
  for (int i = tid * 8; i < CCH * CCH; i += nthr * 8)
    *(v8h*)(dst + i) = *(const v8h*)(src + i);
#endif
}

__device__ __forceinline__ void stage_wait() {
#if USE_ASYNC_LDS
  __builtin_amdgcn_s_wait_asynccnt(0);
#endif
  __syncthreads();
}

// ordered-uint encoding of float for atomic max
__device__ __forceinline__ unsigned f2ord(float f) {
  unsigned u = __float_as_uint(f);
  return (u & 0x80000000u) ? ~u : (u | 0x80000000u);
}
__device__ __forceinline__ float ord2f(unsigned u) {
  return __uint_as_float((u & 0x80000000u) ? (u & 0x7fffffffu) : ~u);
}

// ---------------------------------------------------------------- conversions

__global__ void convert_x_kernel(const float* __restrict__ x,
                                 _Float16* __restrict__ xh, long long n) {
  long long i = (long long)blockIdx.x * blockDim.x + threadIdx.x;
  if (i < n) xh[i] = (_Float16)x[i];
}

// Transpose+convert 6 128x128 matrices into B-fragment layout [mi][n][k].
// mi: 0=Wq 1=Wk 2=Wv 3=root_w 4=pm_w2 5=at_w1
__global__ void convert_w_kernel(const float* __restrict__ Wq, const float* __restrict__ Wk,
                                 const float* __restrict__ Wv, const float* __restrict__ Wr,
                                 const float* __restrict__ pm_w2, const float* __restrict__ at_w1,
                                 _Float16* __restrict__ wt) {
  int i = blockIdx.x * blockDim.x + threadIdx.x;
  if (i >= 6 * CCH * CCH) return;
  int mi = i / (CCH * CCH);
  int r  = i % (CCH * CCH);
  int n  = r / CCH;
  int k  = r % CCH;
  const float* src = (mi == 0) ? Wq : (mi == 1) ? Wk : (mi == 2) ? Wv
                   : (mi == 3) ? Wr : (mi == 4) ? pm_w2 : at_w1;
  wt[i] = (_Float16)src[(size_t)k * CCH + n];   // [k][n] -> [n][k]
}

__global__ void seg_init_kernel(unsigned* __restrict__ smax, float* __restrict__ ssum,
                                long long N) {
  long long i = (long long)blockIdx.x * blockDim.x + threadIdx.x;
  if (i < N) { smax[i] = 0u; ssum[i] = 0.f; }   // 0u == ordered(-inf-ish)
}

// ------------------------------------------------- fused node GEMM: Q|K|V|root
// grid (nodeTiles, 8), block 128 (4 waves). Wave handles col-tile ct = by*4+wave.
__global__ void node_gemm_kernel(const _Float16* __restrict__ xh,
                                 const _Float16* __restrict__ wt,   // 4x [n][k]
                                 const float* __restrict__ root_b,
                                 float* __restrict__ Q, float* __restrict__ Kd,
                                 float* __restrict__ V, float* __restrict__ out,
                                 long long N) {
  __shared__ _Float16 atile[16 * CCH];
  const long long node0 = (long long)blockIdx.x * 16;
  const int t = threadIdx.x;
  { // cooperative A-tile load: 128 threads x 32B = 16x128 f16
    int row = t >> 3;
    int ck  = (t & 7) * 16;
    long long r = node0 + row; if (r >= N) r = N - 1;
    *(v16h*)(atile + row * CCH + ck) = *(const v16h*)(xh + r * CCH + ck);
  }
  __syncthreads();
  const int wave = t >> 5, lane = t & 31;
  const int ct = blockIdx.y * 4 + wave;   // 0..31
  const int mi = ct >> 3;                 // which matrix
  const int nt = ct & 7;                  // 16-col tile inside 128
  const _Float16* w = wt + (size_t)mi * CCH * CCH;
  v8f acc = {};
#pragma unroll
  for (int kt = 0; kt < 4; ++kt)
    acc = wmma_f16(ldsA_frag(atile, lane, kt), glbB_frag(w, lane, nt, kt), acc);
  const int n     = nt * 16 + (lane & 15);
  const int mbase = (lane & 16) ? 8 : 0;
  float* dst = (mi == 0) ? Q : (mi == 1) ? Kd : (mi == 2) ? V : out;
  const float bias = (mi == 3) ? root_b[n] : 0.f;
#pragma unroll
  for (int j = 0; j < 8; ++j) {
    long long r = node0 + mbase + j;
    if (r < N) dst[r * CCH + n] = acc[j] + bias;
  }
}

// ---------------------------------------------------------------- edge kernels

struct EdgeSmem {
  float    dpos[16][3];
  int      rowi[16];
  int      coli[16];
  float    wgt[16];
  _Float16 h1[16 * CCH];
  _Float16 h2[16 * CCH];
};

// Load indices + dpos, compute h1 = relu(dpos @ pm_w1 + pm_b1) into LDS (f16).
__device__ __forceinline__ void edge_h1(EdgeSmem* s, const float* __restrict__ pos,
                                        const long long* __restrict__ ei, long long E,
                                        long long e0, int lane,
                                        const float* __restrict__ pm_w1,
                                        const float* __restrict__ pm_b1) {
  if (lane < 16) {
    long long e = e0 + lane; if (e >= E) e = E - 1;
    long long r = ei[e], c = ei[E + e];
    s->rowi[lane] = (int)r;
    s->coli[lane] = (int)c;
    s->dpos[lane][0] = pos[r * 3 + 0] - pos[c * 3 + 0];
    s->dpos[lane][1] = pos[r * 3 + 1] - pos[c * 3 + 1];
    s->dpos[lane][2] = pos[r * 3 + 2] - pos[c * 3 + 2];
  }
  __syncthreads();
  const int m = lane & 15;
  const int chbase = (lane & 16) ? 64 : 0;
  const float d0 = s->dpos[m][0], d1 = s->dpos[m][1], d2 = s->dpos[m][2];
  for (int c = 0; c < 64; ++c) {
    const int ch = chbase + c;
    float v = d0 * pm_w1[ch] + d1 * pm_w1[CCH + ch] + d2 * pm_w1[2 * CCH + ch] + pm_b1[ch];
    v = v > 0.f ? v : 0.f;
    s->h1[m * CCH + ch] = (_Float16)v;
  }
  __syncthreads();
}

// delta 16x16 tile (nt) = h1 @ pm_w2 + pm_b2, via WMMA over K=128; B from LDS.
__device__ __forceinline__ v8f delta_tile(const EdgeSmem* s,
                                          const _Float16* pm_w2_lds,
                                          const float* __restrict__ pm_b2,
                                          int lane, int nt) {
  v8f acc = {};
#pragma unroll
  for (int kt = 0; kt < 4; ++kt)
    acc = wmma_f16(ldsA_frag(s->h1, lane, kt), ldsB_frag(pm_w2_lds, lane, nt, kt), acc);
  const float b2 = pm_b2[nt * 16 + (lane & 15)];
#pragma unroll
  for (int j = 0; j < 8; ++j) acc[j] += b2;
  return acc;
}

// Pass 1: per-edge attention logits. Block = 64 threads (2 waves), each wave
// owns one 16-edge tile. Weight matrix staged in LDS (async copy overlapped
// with the h1 VALU stage).
__global__ void edge_logits_kernel(const float* __restrict__ pos,
                                   const long long* __restrict__ ei, long long E,
                                   const float* __restrict__ Q, const float* __restrict__ Kd,
                                   const float* __restrict__ pm_w1, const float* __restrict__ pm_b1,
                                   const _Float16* __restrict__ pm_w2t, const float* __restrict__ pm_b2,
                                   const _Float16* __restrict__ at_w1t, const float* __restrict__ at_b1,
                                   const float* __restrict__ at_w2, const float* __restrict__ at_b2,
                                   float* __restrict__ logits) {
  __shared__ _Float16 wbuf[CCH * CCH];   // 32 KB staged weight matrix
  __shared__ EdgeSmem sm[2];
  const int tid = threadIdx.x;
  const int wave = tid >> 5, lane = tid & 31;
  EdgeSmem* s = &sm[wave];
  long long e0 = ((long long)blockIdx.x * 2 + wave) * 16;
  if (e0 + 16 > E) e0 = (E >= 16) ? (E - 16) : 0;

  stage_issue(wbuf, pm_w2t, tid, 64);          // async DMA of pm_w2^T
  edge_h1(s, pos, ei, E, e0, lane, pm_w1, pm_b1);  // overlapped VALU work
  stage_wait();

  // h = Q[col] - K[row] + delta  -> f16 into LDS (A-layout row-major)
  const int mb = (lane & 16) ? 8 : 0;
#pragma unroll 1
  for (int nt = 0; nt < 8; ++nt) {
    v8f d = delta_tile(s, wbuf, pm_b2, lane, nt);
    const int n = nt * 16 + (lane & 15);
#pragma unroll
    for (int j = 0; j < 8; ++j) {
      const int m = mb + j;
      const float h = Q[(size_t)s->coli[m] * CCH + n]
                    - Kd[(size_t)s->rowi[m] * CCH + n] + d[j];
      s->h2[m * CCH + n] = (_Float16)h;
    }
  }
  __syncthreads();                              // h2 complete, wbuf reads done
  stage_issue(wbuf, at_w1t, tid, 64);           // re-stage with at_w1^T
  stage_wait();

  // logits = relu(h @ at_w1 + at_b1) @ at_w2 + at_b2
  float lsum[8];
#pragma unroll
  for (int j = 0; j < 8; ++j) lsum[j] = 0.f;
#pragma unroll 1
  for (int nt = 0; nt < 8; ++nt) {
    v8f acc = {};
#pragma unroll
    for (int kt = 0; kt < 4; ++kt)
      acc = wmma_f16(ldsA_frag(s->h2, lane, kt), ldsB_frag(wbuf, lane, nt, kt), acc);
    const int n = nt * 16 + (lane & 15);
    const float b1v = at_b1[n], w2v = at_w2[n];
#pragma unroll
    for (int j = 0; j < 8; ++j) {
      float r = acc[j] + b1v;
      r = r > 0.f ? r : 0.f;
      float p = r * w2v;
#pragma unroll
      for (int off = 8; off > 0; off >>= 1)   // reduce across the 16 lanes of a half
        p += __shfl_xor(p, off, 16);
      lsum[j] += p;
    }
  }
  if ((lane & 15) == 0) {
    const float b2 = at_b2[0];
#pragma unroll
    for (int j = 0; j < 8; ++j) logits[e0 + mb + j] = lsum[j] + b2;
  }
}

// Pass 2a: segment max over destination nodes (ordered-uint atomic max).
__global__ void seg_max_kernel(const float* __restrict__ logits,
                               const long long* __restrict__ ei, long long E,
                               unsigned* __restrict__ smax) {
  long long e = (long long)blockIdx.x * blockDim.x + threadIdx.x;
  if (e < E) {
    int c = (int)ei[E + e];
    atomicMax(&smax[c], f2ord(logits[e]));
  }
}

// Pass 2b: ew = exp(logit - segmax[col]); segsum[col] += ew
__global__ void edge_exp_kernel(const float* __restrict__ logits,
                                const long long* __restrict__ ei, long long E,
                                const unsigned* __restrict__ smax,
                                float* __restrict__ ssum, float* __restrict__ ew) {
  long long e = (long long)blockIdx.x * blockDim.x + threadIdx.x;
  if (e < E) {
    int c = (int)ei[E + e];
    float w = __expf(logits[e] - ord2f(smax[c]));
    ew[e] = w;
    atomicAdd(&ssum[c], w);
  }
}

// Pass 3: recompute delta (WMMA), weight messages, scatter-add into out.
__global__ void edge_message_kernel(const float* __restrict__ pos,
                                    const long long* __restrict__ ei, long long E,
                                    const float* __restrict__ V,
                                    const float* __restrict__ pm_w1, const float* __restrict__ pm_b1,
                                    const _Float16* __restrict__ pm_w2t, const float* __restrict__ pm_b2,
                                    const float* __restrict__ ew, const float* __restrict__ ssum,
                                    float* __restrict__ out) {
  __shared__ _Float16 wbuf[CCH * CCH];
  __shared__ EdgeSmem sm[2];
  const int tid = threadIdx.x;
  const int wave = tid >> 5, lane = tid & 31;
  EdgeSmem* s = &sm[wave];
  long long e0 = ((long long)blockIdx.x * 2 + wave) * 16;
  if (e0 + 16 > E) e0 = (E >= 16) ? (E - 16) : 0;

  stage_issue(wbuf, pm_w2t, tid, 64);
  edge_h1(s, pos, ei, E, e0, lane, pm_w1, pm_b1);

  if (lane < 16) {
    long long e = e0 + lane; if (e >= E) e = E - 1;
    s->wgt[lane] = ew[e] / ssum[s->coli[lane]];
  }
  stage_wait();

  const int mb = (lane & 16) ? 8 : 0;
#pragma unroll 1
  for (int nt = 0; nt < 8; ++nt) {
    v8f d = delta_tile(s, wbuf, pm_b2, lane, nt);
    const int n = nt * 16 + (lane & 15);
#pragma unroll
    for (int j = 0; j < 8; ++j) {
      const int m = mb + j;
      const float msg = (V[(size_t)s->rowi[m] * CCH + n] + d[j]) * s->wgt[m];
      atomicAdd(&out[(size_t)s->coli[m] * CCH + n], msg);
    }
  }
}

// ------------------------------------------------------------------ launch

extern "C" void kernel_launch(void* const* d_in, const int* in_sizes, int n_in,
                              void* d_out, int out_size, void* d_ws, size_t ws_size,
                              hipStream_t stream) {
  const float*     x      = (const float*)d_in[0];
  const float*     pos    = (const float*)d_in[1];
  const long long* ei     = (const long long*)d_in[2];
  const float*     Wq     = (const float*)d_in[3];
  const float*     Wk     = (const float*)d_in[4];
  const float*     Wv     = (const float*)d_in[5];
  const float*     pm_w1  = (const float*)d_in[6];
  const float*     pm_b1  = (const float*)d_in[7];
  const float*     pm_w2  = (const float*)d_in[8];
  const float*     pm_b2  = (const float*)d_in[9];
  const float*     at_w1  = (const float*)d_in[10];
  const float*     at_b1  = (const float*)d_in[11];
  const float*     at_w2  = (const float*)d_in[12];
  const float*     at_b2  = (const float*)d_in[13];
  const float*     root_w = (const float*)d_in[14];
  const float*     root_b = (const float*)d_in[15];
  float*           out    = (float*)d_out;

  const long long N = in_sizes[0] / CCH;
  const long long E = in_sizes[2] / 2;

  char* ws = (char*)d_ws;
  auto al = [](size_t v) { return (v + 255) & ~(size_t)255; };
  size_t off = 0;
  _Float16* xh   = (_Float16*)(ws + off); off = al(off + (size_t)N * CCH * 2);
  _Float16* wt   = (_Float16*)(ws + off); off = al(off + (size_t)6 * CCH * CCH * 2);
  float*    Q    = (float*)(ws + off);    off = al(off + (size_t)N * CCH * 4);
  float*    Kd   = (float*)(ws + off);    off = al(off + (size_t)N * CCH * 4);
  float*    V    = (float*)(ws + off);    off = al(off + (size_t)N * CCH * 4);
  float*    logi = (float*)(ws + off);    off = al(off + (size_t)E * 4);
  float*    ewb  = (float*)(ws + off);    off = al(off + (size_t)E * 4);
  unsigned* smax = (unsigned*)(ws + off); off = al(off + (size_t)N * 4);
  float*    ssum = (float*)(ws + off);    off = al(off + (size_t)N * 4);

  const _Float16* pm_w2t = wt + (size_t)4 * CCH * CCH;
  const _Float16* at_w1t = wt + (size_t)5 * CCH * CCH;

  // 1) conversions / init
  {
    long long tot = N * CCH;
    convert_x_kernel<<<(unsigned)((tot + 255) / 256), 256, 0, stream>>>(x, xh, tot);
  }
  convert_w_kernel<<<(6 * CCH * CCH + 255) / 256, 256, 0, stream>>>(
      Wq, Wk, Wv, root_w, pm_w2, at_w1, wt);
  seg_init_kernel<<<(unsigned)((N + 255) / 256), 256, 0, stream>>>(smax, ssum, N);

  // 2) fused node GEMM (Q | K | V | root+bias -> out)
  {
    dim3 grid((unsigned)((N + 15) / 16), 8);
    node_gemm_kernel<<<grid, 128, 0, stream>>>(xh, wt, root_b, Q, Kd, V, out, N);
  }

  // 3) edge logits
  {
    long long tiles = (E + 15) / 16;
    unsigned blocks = (unsigned)((tiles + 1) / 2);
    edge_logits_kernel<<<blocks, 64, 0, stream>>>(
        pos, ei, E, Q, Kd, pm_w1, pm_b1, pm_w2t, pm_b2,
        at_w1t, at_b1, at_w2, at_b2, logi);
  }

  // 4) edge softmax (segment max, then exp + segment sum)
  seg_max_kernel<<<(unsigned)((E + 255) / 256), 256, 0, stream>>>(logi, ei, E, smax);
  edge_exp_kernel<<<(unsigned)((E + 255) / 256), 256, 0, stream>>>(logi, ei, E, smax, ssum, ewb);

  // 5) weighted messages scatter-add
  {
    long long tiles = (E + 15) / 16;
    unsigned blocks = (unsigned)((tiles + 1) / 2);
    edge_message_kernel<<<blocks, 64, 0, stream>>>(
        pos, ei, E, V, pm_w1, pm_b1, pm_w2t, pm_b2, ewb, ssum, out);
  }
}
